// EncoderLayer_83193516523785
// MI455X (gfx1250) — compile-verified, matrix operand order checked by reading, not run on hardware
//
#include <hip/hip_runtime.h>

// ---------------------------------------------------------------------------
// MI455X (gfx1250) transformer encoder layer.
// All matmuls via v_wmma_f32_16x16x32_bf16 (bf16 in, fp32 accumulate).
// B=2 S=2048 D=512 H=8 DK=64 DF=2048 -> M = B*S = 4096 rows.
// Weights stored pre-transposed Bt[N][K] bf16 -> all WMMA fragments load as
// contiguous 16B chunks (2x ds_load_b128).  Attention computes S and O
// transposed so softmax reductions are in-lane and P feeds the PV WMMA
// directly from registers.  Tile staging uses gfx1250 async global->LDS
// copies (ASYNCcnt) with double-buffered LDS.
// ---------------------------------------------------------------------------

typedef __attribute__((ext_vector_type(16))) __bf16 v16bf;
typedef __attribute__((ext_vector_type(8)))  float  v8f;

#define MR   4096   // B*S
#define DD   512    // model dim (= H*DK)
#define DFF  2048
#define SS   2048
#define HH   8

__device__ __forceinline__ __bf16 f2bf(float f) {
    union { float f; unsigned u; } c; c.f = f;
    unsigned r = c.u + 0x7FFFu + ((c.u >> 16) & 1u);   // round-to-nearest-even
    unsigned short h = (unsigned short)(r >> 16);
    return __builtin_bit_cast(__bf16, h);
}

__device__ __forceinline__ v8f zero8() {
    v8f z = {0.f, 0.f, 0.f, 0.f, 0.f, 0.f, 0.f, 0.f};
    return z;
}

// --- async global->LDS copy (gfx1250 GLOBAL_LOAD_ASYNC_TO_LDS_B128) --------
#if __has_builtin(__builtin_amdgcn_global_load_async_to_lds_b128)
#define HAS_ASYNC 1
typedef int v4i_ __attribute__((vector_size(16)));
typedef __attribute__((address_space(1))) v4i_* gas_p;
typedef __attribute__((address_space(3))) v4i_* las_p;
__device__ __forceinline__ void cp16(void* lds, const void* g) {
    __builtin_amdgcn_global_load_async_to_lds_b128((gas_p)g, (las_p)lds, 0, 0);
}
__device__ __forceinline__ void wait_cp() {
    asm volatile("s_wait_asynccnt 0x0" ::: "memory");
}
#else
#define HAS_ASYNC 0
__device__ __forceinline__ void cp16(void* lds, const void* g) {
    *reinterpret_cast<uint4*>(lds) = *reinterpret_cast<const uint4*>(g);
}
__device__ __forceinline__ void wait_cp() {}
#endif

// ISA 7.12.2 (16-bit operands, 16x32/32x16): VGPR j<4 -> K=2j+8*half,
// j>=4 -> K=16+2(j-4)+8*half.  All 16 elements of a lane's fragment come from
// one row; the two j-blocks are contiguous 16B chunks -> 2x ds_load_b128.
__device__ __forceinline__ v16bf frag_row(const __bf16* __restrict__ row, int half) {
    v16bf a;
#pragma unroll
    for (int j = 0; j < 8; ++j) {
        int k = (j < 4 ? 2 * j : 16 + 2 * (j - 4)) + 8 * half;
        a[2 * j]     = row[k];
        a[2 * j + 1] = row[k + 1];
    }
    return a;
}

// ---------------------------------------------------------------------------
// bf16 GEMM: C[M,N] = A[M,K] * Bt[N,K]^T  (+bias +resid, relu, q-scale).
// Block tile 128x128x32, 256 threads = 8 waves, wave tile 64x32 (4x2 WMMA).
// LDS rows padded to 40 bf16 (80B: 16B aligned, bank-stride 20 -> clean).
// Double-buffered LDS: async copies fill buf^1 while WMMAs consume buf.
// ---------------------------------------------------------------------------
#define GBM 128
#define GBN 128
#define GBK 32
#define GLD 40

template<bool BIAS, bool RESID, bool RELU, bool OUTF, bool OUTB, bool TROUT, bool SCALE>
__global__ __launch_bounds__(256)
void gemm_bf16_kernel(const __bf16* __restrict__ A, const __bf16* __restrict__ Bt,
                      const float* __restrict__ bias, const float* __restrict__ resid,
                      float* __restrict__ outF, __bf16* __restrict__ outB,
                      int M, int N, int K)
{
    __shared__ __bf16 smA[2][GBM][GLD];
    __shared__ __bf16 smB[2][GBN][GLD];

    const int tid  = threadIdx.x;
    const int lane = tid & 31;
    const int w    = tid >> 5;
    const int half = (lane >> 4) & 1;
    const int l16  = lane & 15;
    const int wm   = w >> 2;      // 0..1  -> 64 rows
    const int wn   = w & 3;       // 0..3  -> 32 cols
    const int m0   = blockIdx.y * GBM;
    const int n0   = blockIdx.x * GBN;

    v8f acc[4][2];
#pragma unroll
    for (int mi = 0; mi < 4; ++mi)
#pragma unroll
        for (int ni = 0; ni < 2; ++ni) acc[mi][ni] = zero8();

    const int r   = tid >> 1;
    const int c16 = (tid & 1) * 16;
    const __bf16* aSrc = A  + (size_t)(m0 + r) * K + c16;
    const __bf16* bSrc = Bt + (size_t)(n0 + r) * K + c16;

    auto stage = [&](int buf, int kt) {
        cp16(&smA[buf][r][c16],     aSrc + kt);
        cp16(&smA[buf][r][c16 + 8], aSrc + kt + 8);
        cp16(&smB[buf][r][c16],     bSrc + kt);
        cp16(&smB[buf][r][c16 + 8], bSrc + kt + 8);
    };

    stage(0, 0);
    int cur = 0;
    for (int kt = 0; kt < K; kt += GBK) {
        wait_cp();            // this wave's async copies done
        __syncthreads();      // all waves' copies landed; prev buffer free
        if (kt + GBK < K) stage(cur ^ 1, kt + GBK);

        v16bf af[4], bfr[2];
#pragma unroll
        for (int mi = 0; mi < 4; ++mi)
            af[mi] = frag_row(&smA[cur][wm * 64 + mi * 16 + l16][0], half);
#pragma unroll
        for (int ni = 0; ni < 2; ++ni)
            bfr[ni] = frag_row(&smB[cur][wn * 32 + ni * 16 + l16][0], half);

#pragma unroll
        for (int mi = 0; mi < 4; ++mi)
#pragma unroll
            for (int ni = 0; ni < 2; ++ni)
                acc[mi][ni] = __builtin_amdgcn_wmma_f32_16x16x32_bf16(
                    false, af[mi], false, bfr[ni], (short)0, acc[mi][ni], false, false);

        cur ^= 1;
    }

    // Epilogue: C layout lane = column (within 16), VGPR i = row i + 8*half.
#pragma unroll
    for (int mi = 0; mi < 4; ++mi) {
#pragma unroll
        for (int ni = 0; ni < 2; ++ni) {
            const int col = n0 + wn * 32 + ni * 16 + l16;
            float bval = 0.f;
            if constexpr (BIAS) bval = bias[col];
#pragma unroll
            for (int i = 0; i < 8; ++i) {
                const int row = m0 + wm * 64 + mi * 16 + i + 8 * half;
                const size_t idx = (size_t)row * N + col;
                float v = acc[mi][ni][i] + bval;
                if constexpr (RESID) v += resid[idx];
                if constexpr (RELU)  v = fmaxf(v, 0.f);
                if constexpr (SCALE) v *= 0.125f;     // 1/sqrt(DK), exact in bf16
                if constexpr (OUTF)  outF[idx] = v;
                if constexpr (OUTB) {
                    if constexpr (TROUT) outB[(size_t)col * M + row] = f2bf(v);
                    else                 outB[idx] = f2bf(v);
                }
            }
        }
    }
}

// ---------------------------------------------------------------------------
// Flash attention, transposed-score formulation.
// One block = (b, h, 64 query rows); 4 waves, 16 queries each.
//   S^T(keys x queries) = K_tile * Q^T   (A = K rows, B = Q rows)
//   -> lane = query, VGPR = key: softmax reductions are in-lane + 1 xor16.
//   P^T lives in registers already laid out as the next WMMA's B operand.
//   O^T(vdim x queries) = V^T * P^T      (A = Vt rows from LDS, B = p regs)
//   -> rescale is a per-lane scalar; store 8 consecutive bf16 per tile (b128).
// Q is pre-scaled by 1/sqrt(DK) in the Q-projection GEMM.
// K/V chunks streamed via async global->LDS into a double buffer.
// ---------------------------------------------------------------------------
#define ALDK 72   // smK row stride: 144B (16B aligned, bank-stride 36 -> clean)

__global__ __launch_bounds__(128)
void attn_kernel(const __bf16* __restrict__ Q, const __bf16* __restrict__ Kb,
                 const __bf16* __restrict__ Vt, __bf16* __restrict__ O)
{
    __shared__ __bf16 smK[2][32][ALDK];    // [key][kdim 0..63]
    __shared__ __bf16 smV[2][64][GLD];     // [vdim][key 0..31]  (transposed V)

    const int tid  = threadIdx.x;
    const int lane = tid & 31;
    const int w    = tid >> 5;
    const int half = (lane >> 4) & 1;
    const int l16  = lane & 15;
    const int mt   = blockIdx.x;
    const int h    = blockIdx.y;
    const int b    = blockIdx.z;
    const size_t baserow = (size_t)b * SS;

    // Per-lane query row; fragments double as the B operand of S^T = K*Q^T.
    const int qrow = mt * 64 + w * 16 + l16;
    const __bf16* qptr = Q + (baserow + qrow) * DD + h * 64;
    const v16bf qa0 = frag_row(qptr, half);
    const v16bf qa1 = frag_row(qptr + 32, half);

    v8f o[4];                 // O^T tiles: lane = query, VGPR = vdim i+8*half
#pragma unroll
    for (int nt = 0; nt < 4; ++nt) o[nt] = zero8();
    float mq = -1e30f, lq = 0.f;   // per-query online softmax state

    const int kr = tid >> 2, kc = (tid & 3) * 16;   // K tile: 32 x 64
    const int vr = tid >> 1, vc = (tid & 1) * 16;   // Vt tile: 64 x 32
    const __bf16* kSrc = Kb + (baserow + kr) * DD + h * 64 + kc;
    const __bf16* vSrc = Vt + (size_t)(h * 64 + vr) * MR + baserow + vc;

    auto stage = [&](int buf, int t0) {
        cp16(&smK[buf][kr][kc],     kSrc + (size_t)t0 * DD);
        cp16(&smK[buf][kr][kc + 8], kSrc + (size_t)t0 * DD + 8);
        cp16(&smV[buf][vr][vc],     vSrc + t0);
        cp16(&smV[buf][vr][vc + 8], vSrc + t0 + 8);
    };

    stage(0, 0);
    int cur = 0;
    for (int t0 = 0; t0 < SS; t0 += 32) {
        wait_cp();
        __syncthreads();
        if (t0 + 32 < SS) stage(cur ^ 1, t0 + 32);

        // S^T tiles: s0 = keys 0..15 (x 16 queries), s1 = keys 16..31.
        v8f s0 = zero8(), s1 = zero8();
        {
            const __bf16* k0 = &smK[cur][l16][0];        // A-lane = key row
            const __bf16* k1 = &smK[cur][16 + l16][0];
            s0 = __builtin_amdgcn_wmma_f32_16x16x32_bf16(false, frag_row(k0, half),      false, qa0, (short)0, s0, false, false);
            s0 = __builtin_amdgcn_wmma_f32_16x16x32_bf16(false, frag_row(k0 + 32, half), false, qa1, (short)0, s0, false, false);
            s1 = __builtin_amdgcn_wmma_f32_16x16x32_bf16(false, frag_row(k1, half),      false, qa0, (short)0, s1, false, false);
            s1 = __builtin_amdgcn_wmma_f32_16x16x32_bf16(false, frag_row(k1 + 32, half), false, qa1, (short)0, s1, false, false);
        }

        // In-lane max over this lane's 16 keys, combine with partner lane.
        float vmax = s0[0];
#pragma unroll
        for (int i = 1; i < 8; ++i) vmax = fmaxf(vmax, s0[i]);
#pragma unroll
        for (int i = 0; i < 8; ++i) vmax = fmaxf(vmax, s1[i]);
        vmax = fmaxf(vmax, __shfl_xor(vmax, 16));

        const float mn = fmaxf(mq, vmax);
        const float rescale = __expf(mq - mn);
        mq = mn;

        // p = exp(s - mn); this lane's keys {kmap(j,half)} are exactly the
        // B-fragment pattern for the PV WMMA -> pack directly into registers.
        v16bf pb;
        float psum = 0.f;
#pragma unroll
        for (int i = 0; i < 8; ++i) {
            const float p0 = __expf(s0[i] - mn);
            const float p1 = __expf(s1[i] - mn);
            pb[i]     = f2bf(p0);
            pb[8 + i] = f2bf(p1);
            psum += p0 + p1;
        }
        psum += __shfl_xor(psum, 16);
        lq = lq * rescale + psum;

#pragma unroll
        for (int nt = 0; nt < 4; ++nt) {
#pragma unroll
            for (int i = 0; i < 8; ++i) o[nt][i] *= rescale;
            o[nt] = __builtin_amdgcn_wmma_f32_16x16x32_bf16(
                false, frag_row(&smV[cur][nt * 16 + l16][0], half), false, pb,
                (short)0, o[nt], false, false);
        }

        cur ^= 1;
    }

    // O^T: lane = query, VGPR i = vdim nt*16 + i + 8*half -> 8 consecutive
    // bf16 per tile = one 16B store; concat-head layout feeds the Wo GEMM.
    const float inv = 1.f / lq;
    __bf16* orow = O + (baserow + qrow) * DD + h * 64;
#pragma unroll
    for (int nt = 0; nt < 4; ++nt) {
        union { uint4 u; __bf16 e[8]; } pk;
#pragma unroll
        for (int i = 0; i < 8; ++i) pk.e[i] = f2bf(o[nt][i] * inv);
        *reinterpret_cast<uint4*>(orow + nt * 16 + 8 * half) = pk.u;
    }
}

// ---------------------------------------------------------------------------
// LayerNorm over D=512: one block (128 threads) per row.
// ---------------------------------------------------------------------------
__global__ __launch_bounds__(128)
void ln_kernel(const float* __restrict__ a, const float* __restrict__ g,
               const float* __restrict__ be, float* __restrict__ outF,
               __bf16* __restrict__ outB)
{
    const int row = blockIdx.x;
    const int tid = threadIdx.x;
    const float* ar = a + (size_t)row * DD;

    float v[4], s = 0.f, s2 = 0.f;
#pragma unroll
    for (int j = 0; j < 4; ++j) {
        const float x = ar[tid + 128 * j];
        v[j] = x; s += x; s2 += x * x;
    }
#pragma unroll
    for (int off = 16; off >= 1; off >>= 1) {
        s  += __shfl_xor(s, off);
        s2 += __shfl_xor(s2, off);
    }
    __shared__ float red[2][4];
    const int w = tid >> 5;
    if ((tid & 31) == 0) { red[0][w] = s; red[1][w] = s2; }
    __syncthreads();
    s  = red[0][0] + red[0][1] + red[0][2] + red[0][3];
    s2 = red[1][0] + red[1][1] + red[1][2] + red[1][3];

    const float mu = s * (1.f / DD);
    const float var = s2 * (1.f / DD) - mu * mu;
    const float rs = rsqrtf(var + 1e-5f);
#pragma unroll
    for (int j = 0; j < 4; ++j) {
        const int c = tid + 128 * j;
        const float o = (v[j] - mu) * rs * g[c] + be[c];
        outF[(size_t)row * DD + c] = o;
        if (outB) outB[(size_t)row * DD + c] = f2bf(o);
    }
}

// ---------------------------------------------------------------------------
// Conversions: fp32 -> bf16 (plain, transposed, and QKV-weight reshape+T).
// ---------------------------------------------------------------------------
__global__ __launch_bounds__(256)
void cvt_kernel(const float* __restrict__ in, __bf16* __restrict__ out, int n)
{
    const int i = blockIdx.x * 256 + threadIdx.x;
    if (i < n) out[i] = f2bf(in[i]);
}

// in: [K][N] fp32 -> out: [N][K] bf16 (coalesced read, 2B scattered write).
__global__ __launch_bounds__(256)
void cvt_t_kernel(const float* __restrict__ in, __bf16* __restrict__ out, int K, int N)
{
    const int i = blockIdx.x * 256 + threadIdx.x;
    if (i >= K * N) return;
    const int k = i / N, n = i - k * N;
    out[(size_t)n * K + k] = f2bf(in[i]);
}

// Wq/Wk/Wv: [H, D, DK] -> Bt [H*DK, D] bf16 (GEMM K dim = D).
__global__ __launch_bounds__(256)
void cvt_qkvw_t_kernel(const float* __restrict__ in, __bf16* __restrict__ out)
{
    const int i = blockIdx.x * 256 + threadIdx.x;   // n = H*D*DK = 262144
    if (i >= HH * DD * 64) return;
    const int k = i & 63;              // dk
    const int d = (i >> 6) & (DD - 1); // d
    const int h = i >> 15;             // head
    out[(size_t)(h * 64 + k) * DD + d] = f2bf(in[i]);
}

// ---------------------------------------------------------------------------
extern "C" void kernel_launch(void* const* d_in, const int* in_sizes, int n_in,
                              void* d_out, int out_size, void* d_ws, size_t ws_size,
                              hipStream_t stream)
{
    (void)in_sizes; (void)n_in; (void)out_size; (void)ws_size;

    const float* x   = (const float*)d_in[0];
    const float* Wq  = (const float*)d_in[1];
    const float* bq  = (const float*)d_in[2];
    const float* Wk  = (const float*)d_in[3];
    const float* bk  = (const float*)d_in[4];
    const float* Wv  = (const float*)d_in[5];
    const float* bv  = (const float*)d_in[6];
    const float* Wo  = (const float*)d_in[7];
    const float* bo  = (const float*)d_in[8];
    const float* g1  = (const float*)d_in[9];
    const float* be1 = (const float*)d_in[10];
    const float* W1  = (const float*)d_in[11];
    const float* b1  = (const float*)d_in[12];
    const float* W2  = (const float*)d_in[13];
    const float* b2  = (const float*)d_in[14];
    const float* g2  = (const float*)d_in[15];
    const float* be2 = (const float*)d_in[16];

    size_t off = 0;
    auto take = [&](size_t bytes) -> void* {
        void* p = (char*)d_ws + off;
        off += (bytes + 255) & ~(size_t)255;
        return p;
    };

    __bf16* xb   = (__bf16*)take((size_t)MR * DD * 2);
    __bf16* wqbt = (__bf16*)take((size_t)DD * DD * 2);
    __bf16* wkbt = (__bf16*)take((size_t)DD * DD * 2);
    __bf16* wvbt = (__bf16*)take((size_t)DD * DD * 2);
    __bf16* wobt = (__bf16*)take((size_t)DD * DD * 2);
    __bf16* w1bt = (__bf16*)take((size_t)DD * DFF * 2);
    __bf16* w2bt = (__bf16*)take((size_t)DFF * DD * 2);
    __bf16* qb   = (__bf16*)take((size_t)MR * DD * 2);   // dead after attention
    __bf16* kbb  = (__bf16*)take((size_t)MR * DD * 2);   // dead after attention
    __bf16* vt   = (__bf16*)take((size_t)MR * DD * 2);   // [DD][MR]; dead after attn
    __bf16* ab   = (__bf16*)take((size_t)MR * DD * 2);   // dead after Wo GEMM
    float*  sum  = (float*)take((size_t)MR * DD * 4);    // sum1, reused as sum2
    float*  hf   = (float*)take((size_t)MR * DD * 4);
    __bf16* hb   = (__bf16*)take((size_t)MR * DD * 2);
    __bf16* ff1  = qb;   // reuse qb..ab (exactly 16 MB) for FFN hidden [MR, DFF]

    // --- conversions (one-time, bandwidth-trivial) ---
    cvt_kernel<<<(MR * DD + 255) / 256, 256, 0, stream>>>(x, xb, MR * DD);
    cvt_qkvw_t_kernel<<<(HH * DD * 64 + 255) / 256, 256, 0, stream>>>(Wq, wqbt);
    cvt_qkvw_t_kernel<<<(HH * DD * 64 + 255) / 256, 256, 0, stream>>>(Wk, wkbt);
    cvt_qkvw_t_kernel<<<(HH * DD * 64 + 255) / 256, 256, 0, stream>>>(Wv, wvbt);
    cvt_t_kernel<<<(DD * DD + 255) / 256, 256, 0, stream>>>(Wo, wobt, DD, DD);
    cvt_t_kernel<<<(DD * DFF + 255) / 256, 256, 0, stream>>>(W1, w1bt, DD, DFF);
    cvt_t_kernel<<<(DFF * DD + 255) / 256, 256, 0, stream>>>(W2, w2bt, DFF, DD);

    const dim3 g512(DD / GBN, MR / GBM);    // (4, 32)
    const dim3 g2048(DFF / GBN, MR / GBM);  // (16, 32)

    // --- QKV projections (bias fused; Q pre-scaled; V written transposed) ---
    gemm_bf16_kernel<true, false, false, false, true, false, true>
        <<<g512, 256, 0, stream>>>(xb, wqbt, bq, nullptr, nullptr, qb, MR, DD, DD);
    gemm_bf16_kernel<true, false, false, false, true, false, false>
        <<<g512, 256, 0, stream>>>(xb, wkbt, bk, nullptr, nullptr, kbb, MR, DD, DD);
    gemm_bf16_kernel<true, false, false, false, true, true, false>
        <<<g512, 256, 0, stream>>>(xb, wvbt, bv, nullptr, nullptr, vt, MR, DD, DD);

    // --- attention (flash, transposed-score online softmax) ---
    attn_kernel<<<dim3(SS / 64, HH, 2), 128, 0, stream>>>(qb, kbb, vt, ab);

    // --- output projection + residual 1 ---
    gemm_bf16_kernel<true, true, false, true, false, false, false>
        <<<g512, 256, 0, stream>>>(ab, wobt, bo, x, sum, nullptr, MR, DD, DD);
    ln_kernel<<<MR, 128, 0, stream>>>(sum, g1, be1, hf, hb);

    // --- FFN ---
    gemm_bf16_kernel<true, false, true, false, true, false, false>
        <<<g2048, 256, 0, stream>>>(hb, w1bt, b1, nullptr, nullptr, ff1, MR, DFF, DD);
    gemm_bf16_kernel<true, true, false, true, false, false, false>
        <<<g512, 256, 0, stream>>>(ff1, w2bt, b2, hf, sum, nullptr, MR, DD, DFF);
    ln_kernel<<<MR, 128, 0, stream>>>(sum, g2, be2, (float*)d_out, nullptr);
}
